// CycleFC_11828339933672
// MI455X (gfx1250) — matrix operand order repaired and checked by reading, not compiled, not run on hardware
//
#include <hip/hip_runtime.h>
#include <hip/hip_bf16.h>

typedef __attribute__((ext_vector_type(16))) __bf16 v16bf;
typedef __attribute__((ext_vector_type(8)))  __bf16 v8bf;
typedef __attribute__((ext_vector_type(8)))  float  v8f;

#define CB 32      // batch
#define CC 256     // in channels (= K)
#define CO 256     // out channels (= N of GEMM)
#define CH 56
#define CW 56
#define CHW 3136   // 56*56, = 98 * 32
#define MT 32      // spatial tile per block
#define KT 32      // K per WMMA step
#define NKK (CC / KT)
#define LDA 40     // padded LDS row stride (halfs) for A tile
#define LDO 36     // padded LDS row stride (floats) for out transpose (16B aligned)

// ---------------------------------------------------------------------------
// Prepack: w[O][C] f32 -> wpk[kk][o][k32] bf16 so each WMMA lane reads its
// 16 B-fragment values as one aligned 32B chunk.
// ---------------------------------------------------------------------------
__global__ __launch_bounds__(256) void cyclefc_prepack(const float* __restrict__ wgt,
                                                       __bf16* __restrict__ wpk) {
    int gid = blockIdx.x * 256 + threadIdx.x;       // 0 .. 65535
    int k32 = gid & 31;
    int o   = (gid >> 5) & 255;
    int kk  = gid >> 13;                            // 0..7
    wpk[gid] = (__bf16)wgt[o * CC + kk * KT + k32];
}

// ---------------------------------------------------------------------------
// Main kernel: one block = 32 spatial rows x all 256 output channels.
// 8 waves: (msub in {0,1}) x (group of 4 o-tiles). Double-buffered A tile.
// ---------------------------------------------------------------------------
template <bool PACKED>
__global__ __launch_bounds__(256) void cyclefc_wmma(const float* __restrict__ x,
                                                    const float* __restrict__ wgt,
                                                    const __bf16* __restrict__ wpk,
                                                    const float* __restrict__ bias,
                                                    float* __restrict__ out) {
    __shared__ __align__(16) __bf16 ldsA[2][MT * LDA];  // 2 x 2.5 KB shifted A tile
    __shared__ __align__(16) float ldsOut[CO * LDO];    // 36 KB transpose buffer

    const int t    = threadIdx.x;
    const int lane = t & 31;
    const int wave = t >> 5;
    const int b    = blockIdx.y;
    const int n0   = blockIdx.x * MT;                   // spatial tile start (flat h*W+w)

    const int msub = wave & 1;                          // which 16-row M subtile
    const int og   = wave >> 1;                         // which group of 4 o-tiles
    const int hi   = lane >> 4;                         // K-half select per A/B layout
    const int nn   = lane & 15;                         // M row (A) / N col (B,D)

    // cooperative A-gather: 8 threads x 4 m cover 32 m for one channel
    const int kLoc  = t >> 3;                           // 0..31: channel within K tile
    const int mBase = (t & 7) * 4;

    // Hoisted per-thread spatial coords for its 4 gather rows (kk-invariant).
    int rowb[4], wcol[4];
    {
        int n  = n0 + mBase;
        int hh = n / CW;
        int ww = n - hh * CW;
#pragma unroll
        for (int i = 0; i < 4; ++i) {
            rowb[i] = hh * CW;
            wcol[i] = ww;
            ++ww;
            int carry = (ww == CW);
            hh += carry;
            ww = carry ? 0 : ww;
        }
    }

    // Block-uniform base => saddr + 32-bit voffset addressing for the gather.
    const float* __restrict__ xb = x + (size_t)(b * CC) * CHW;
    unsigned chanOff = (unsigned)(kLoc * CHW);          // running c*CHW (elements)
    int shr = (kLoc + 3) % 7;                           // running (c+3)%7; +32ch => +4 mod 7

    // Branch-free shifted gather (must be called in kk order: 0,1,...).
    auto gather = [&](float g[4]) {
        const int d = shr - 3;                          // per-channel width shift
        shr += 4; shr = (shr >= 7) ? shr - 7 : shr;     // advance residue, no division
#pragma unroll
        for (int i = 0; i < 4; ++i) {
            int  wp    = wcol[i] + d;
            bool valid = (unsigned)wp < (unsigned)CW;
            unsigned off = chanOff + (unsigned)(rowb[i] + (valid ? wp : 0));
            float v = xb[off];                          // unconditional, always in-plane
            g[i] = valid ? v : 0.0f;                    // cndmask, no exec-mask branch
        }
        chanOff += (unsigned)(KT * CHW);
    };
    auto stage = [&](int buf, const float g[4]) {
        __bf16* dst = &ldsA[buf][mBase * LDA + kLoc];
#pragma unroll
        for (int i = 0; i < 4; ++i) dst[i * LDA] = (__bf16)g[i];
    };

    v8f acc[4] = {};

    // Prologue: stage K-step 0 into buffer 0.
    {
        float g[4];
        gather(g);
        stage(0, g);
    }

    for (int kk = 0; kk < NKK; ++kk) {
        const int buf = kk & 1;
        __syncthreads();                                // ldsA[buf] visible to all

        // Kick off next K-step's gather loads; latency hides under WMMAs below.
        float gnext[4];
        if (kk + 1 < NKK) gather(gnext);

        // A fragment: contiguous K order. lane(lo): K 0..15; lane(hi): K 16..31
        const __bf16* arow = &ldsA[buf][(msub * 16 + nn) * LDA];
        v8bf alo = *(const v8bf*)(arow + hi * 16);
        v8bf ahi = *(const v8bf*)(arow + hi * 16 + 8);
        v16bf afrag;
#pragma unroll
        for (int i = 0; i < 8; ++i) { afrag[i] = alo[i]; afrag[i + 8] = ahi[i]; }

        const int c0 = kk * KT;
#pragma unroll
        for (int tt = 0; tt < 4; ++tt) {
            const int o = (og * 4 + tt) * 16 + nn;      // output channel = B column
            v16bf bfrag;
            if (PACKED) {
                const unsigned boff = (unsigned)((kk * CO + o) * KT + hi * 16);
                v8bf b0 = *(const v8bf*)(wpk + boff);
                v8bf b1 = *(const v8bf*)(wpk + boff + 8);
#pragma unroll
                for (int i = 0; i < 8; ++i) { bfrag[i] = b0[i]; bfrag[i + 8] = b1[i]; }
            } else {
                const float* wrow = wgt + o * CC + c0 + hi * 16;
                __align__(16) float wtmp[16];
                *(float4*)&wtmp[0]  = *(const float4*)(wrow + 0);
                *(float4*)&wtmp[4]  = *(const float4*)(wrow + 4);
                *(float4*)&wtmp[8]  = *(const float4*)(wrow + 8);
                *(float4*)&wtmp[12] = *(const float4*)(wrow + 12);
#pragma unroll
                for (int i = 0; i < 16; ++i) bfrag[i] = (__bf16)wtmp[i];
            }
            acc[tt] = __builtin_amdgcn_wmma_f32_16x16x32_bf16(
                false, afrag, false, bfrag, (short)0, acc[tt], false, false);
        }

        // Stage next tile into the other buffer (waits only on its own loads).
        if (kk + 1 < NKK) stage(buf ^ 1, gnext);
    }

    // Epilogue: bias add + LDS transpose so global stores are coalesced b128 rows.
    // C/D layout: lane nn = N, VGPR r -> M = hi*8 + r (within the 16-row subtile).
#pragma unroll
    for (int tt = 0; tt < 4; ++tt) {
        const int o  = (og * 4 + tt) * 16 + nn;
        const float bv = bias[o];
        float* orow = &ldsOut[o * LDO + msub * 16 + hi * 8];   // 16B aligned
#pragma unroll
        for (int r = 0; r < 8; ++r) orow[r] = acc[tt][r] + bv;
    }
    __syncthreads();

    // 8 x float4 per thread: each 8-thread group writes one channel's 32 floats (128B).
    const int ocBase = t >> 3;            // 0..31
    const int m4     = (t & 7) * 4;       // 0,4,...,28
#pragma unroll
    for (int i = 0; i < 8; ++i) {
        const int oc = i * 32 + ocBase;
        float4 vv = *(const float4*)(&ldsOut[oc * LDO + m4]);
        *(float4*)(&out[(size_t)(b * CO + oc) * CHW + n0 + m4]) = vv;
    }
}

extern "C" void kernel_launch(void* const* d_in, const int* in_sizes, int n_in,
                              void* d_out, int out_size, void* d_ws, size_t ws_size,
                              hipStream_t stream) {
    const float* x    = (const float*)d_in[0];
    const float* wgt  = (const float*)d_in[1];
    const float* bias = (const float*)d_in[2];
    float* out = (float*)d_out;

    const size_t packed_bytes = (size_t)NKK * CO * KT * sizeof(__bf16); // 128 KB
    const bool use_packed = (d_ws != nullptr) && (ws_size >= packed_bytes);

    dim3 grid(CHW / MT, CB);   // (98, 32)
    dim3 block(256);

    if (use_packed) {
        __bf16* wpk = (__bf16*)d_ws;
        cyclefc_prepack<<<dim3(256), dim3(256), 0, stream>>>(wgt, wpk);
        cyclefc_wmma<true><<<grid, block, 0, stream>>>(x, wgt, wpk, bias, out);
    } else {
        cyclefc_wmma<false><<<grid, block, 0, stream>>>(x, wgt, (const __bf16*)d_ws, bias, out);
    }
}
